// Linear_57698590654767
// MI455X (gfx1250) — compile-verified, matrix operand order checked by reading, not run on hardware
//
#include <hip/hip_runtime.h>
#include <stdint.h>

typedef __attribute__((ext_vector_type(16))) __bf16 v16bf;
typedef __attribute__((ext_vector_type(8)))  float  v8f;
typedef __attribute__((ext_vector_type(4)))  __bf16 bf16x4;
typedef __attribute__((ext_vector_type(4)))  unsigned int v4u32;
typedef __attribute__((ext_vector_type(8)))  int v8i32;
typedef __attribute__((ext_vector_type(4)))  int v4i32;

#define M_DIM 16384
#define N_DIM 4096
#define K_DIM 4096
#define PITCH 80   // bytes per 32-bf16 LDS row: 64B data + 16B pad (20-bank stride, conflict-free)

// ---------------- W fp32 -> bf16 (feeds TDM path; W_bf16 fits in 192MB L2) ----------------
__global__ __launch_bounds__(256) void wconv_kernel(const float* __restrict__ W,
                                                    __bf16* __restrict__ o) {
    const size_t i = ((size_t)blockIdx.x * 256 + threadIdx.x) * 4;
    const float4 v = *(const float4*)(W + i);
    bf16x4 r;
    r.x = (__bf16)v.x; r.y = (__bf16)v.y; r.z = (__bf16)v.z; r.w = (__bf16)v.w;
    *(bf16x4*)(o + i) = r;
}

// ---------------- h[m,r] = sum_k data[m,k] * lora_a[g(m), r, k]  (rank 16, tiny) -----------
__global__ __launch_bounds__(256) void lora_h_kernel(const float* __restrict__ data,
                                                     const float* __restrict__ lora_a,
                                                     float* __restrict__ h) {
    const int idx = blockIdx.x * 256 + threadIdx.x;   // [0, M_DIM*16)
    const int m = idx >> 4;
    const int r = idx & 15;
    const int g = m >> 12;                            // 4096 tokens per adapter group
    const float4* __restrict__ xp = (const float4*)(data + (size_t)m * K_DIM);
    const float4* __restrict__ ap = (const float4*)(lora_a + ((size_t)(g * 16 + r)) * K_DIM);
    float acc = 0.f;
    for (int k = 0; k < K_DIM / 4; ++k) {
        const float4 x = xp[k], a = ap[k];
        acc += x.x * a.x + x.y * a.y + x.z * a.z + x.w * a.w;
    }
    h[idx] = acc;
}

// ---------------- async fallback (proven to assemble in round 1) ---------------------------
__device__ __forceinline__ void async_b128(uint32_t lds_addr, uint32_t goff, const void* sbase) {
    asm volatile("global_load_async_to_lds_b128 %0, %1, %2"
                 :: "v"(lds_addr), "v"(goff), "s"(sbase) : "memory");
}
__device__ __forceinline__ void wait_async0() {
    asm volatile("s_wait_asynccnt 0x0" ::: "memory");
}

__global__ __launch_bounds__(256) void lora_gemm_kernel(
    const float*  __restrict__ data,
    const float*  __restrict__ bias,
    const float*  __restrict__ lora_b,
    const __bf16* __restrict__ wbf,
    const float*  __restrict__ hbuf,
    float*        __restrict__ out) {

    __shared__ __align__(16) unsigned char shA[2][128 * PITCH];
    __shared__ __align__(16) unsigned char shB[2][128 * PITCH];

    const int tid  = threadIdx.x;
    const int bn   = blockIdx.x;            // N/128 = 32
    const int bm   = blockIdx.y;            // M/128 = 128
    const int m0   = bm * 128, n0 = bn * 128;
    const int wave = tid >> 5, lane = tid & 31;
    const int wm   = wave >> 1;             // 0..3 -> 32-row slab
    const int wn   = wave & 1;              // 0..1 -> 64-col slab

    // staging assignment: 2 threads per tile row (each covers 16 K-elements)
    const int row  = tid >> 1;              // 0..127
    const int half = tid & 1;

    const float* __restrict__ agp = data + (size_t)(m0 + row) * K_DIM + half * 16;
    const uint32_t shB_base = (uint32_t)(uintptr_t)(&shB[0][0]);

    v8f acc[2][4];
#pragma unroll
    for (int i = 0; i < 2; ++i)
#pragma unroll
        for (int j = 0; j < 4; ++j)
            acc[i][j] = (v8f){0.f, 0.f, 0.f, 0.f, 0.f, 0.f, 0.f, 0.f};

    float4 av[4];
    auto loadA = [&](int kk) {
        const float4* p = (const float4*)(agp + kk);
#pragma unroll
        for (int i = 0; i < 4; ++i) av[i] = p[i];
    };
    auto storeA = [&](int buf) {
        union { __bf16 h[16]; uint4 q[2]; } pk;
#pragma unroll
        for (int i = 0; i < 4; ++i) {
            pk.h[4 * i + 0] = (__bf16)av[i].x;
            pk.h[4 * i + 1] = (__bf16)av[i].y;
            pk.h[4 * i + 2] = (__bf16)av[i].z;
            pk.h[4 * i + 3] = (__bf16)av[i].w;
        }
        uint4* dst = (uint4*)(&shA[buf][row * PITCH + half * 32]);
        dst[0] = pk.q[0];
        dst[1] = pk.q[1];
    };

#if __has_builtin(__builtin_amdgcn_tensor_load_to_lds)
    // ---- Tensor Data Mover: one D# per 128x32 bf16 tile; HW applies 16B pad per 64B row ----
    auto tileB = [&](int kk, int buf) {
        const uint64_t gaddr = (uint64_t)(uintptr_t)wbf +
                               ((uint64_t)n0 * K_DIM + (uint64_t)kk) * 2ull;
        const uint32_t lds = shB_base + (uint32_t)(buf * (128 * PITCH));
        v4u32 g0;
        g0.x = 1u;                                      // count=1 valid descriptor
        g0.y = lds;                                     // lds_addr [63:32]
        g0.z = (uint32_t)gaddr;                         // global_addr [95:64]
        g0.w = (uint32_t)(gaddr >> 32) | (2u << 30);    // global_addr[56:32] | type=2
        v8i32 g1;
        g1[0] = (int)((1u << 16)        // data_size = 1 -> 2 bytes
                    | (1u << 20)        // pad_enable
                    | (3u << 22)        // pad_interval = 3 -> 16 DWORDs (64B) between pads
                    | (3u << 25));      // pad_amount   = 3 -> 4 DWORDs (16B) of pad
        g1[1] = (int)(4096u << 16);     // tensor_dim0 = 4096 (bits 79:48, low half)
        g1[2] = (int)(4096u << 16);     // tensor_dim1 = 4096 (bits 111:80, low half)
        g1[3] = (int)(32u << 16);       // tile_dim0 = 32 elements (bits 127:112)
        g1[4] = 128;                    // tile_dim1 = 128 rows; tile_dim2 = 0
        g1[5] = 4096;                   // tensor_dim0_stride = 4096 elements (bits 207:160 lo)
        g1[6] = 0;
        g1[7] = 0;
        const v4i32 z4 = (v4i32)0;      // groups 2/3 unused for 2-D tensor
        const v8i32 z8 = (v8i32)0;      // 6-arg toolchain form: extra zero-filled group
        __builtin_amdgcn_tensor_load_to_lds(g0, g1, z4, z4, z8, 0);
    };
    auto waitB = [&]() { __builtin_amdgcn_s_wait_tensorcnt(0); };
    const bool issuer = (wave == 0);    // wave-uniform: single TDM issue per workgroup
#else
    const uint32_t bgo_base = (uint32_t)(n0 + row) * (K_DIM * 2) + half * 32;
    auto tileB = [&](int kk, int buf) {
        const uint32_t go = bgo_base + (uint32_t)(kk * 2);
        const uint32_t la = shB_base + (uint32_t)(buf * (128 * PITCH) + row * PITCH + half * 32);
        async_b128(la,      go,      (const void*)wbf);
        async_b128(la + 16, go + 16, (const void*)wbf);
    };
    auto waitB = [&]() { wait_async0(); };
    const bool issuer = true;
#endif

    const int fr = lane & 15;
    const int fo = (lane & 16) ? 16 : 0;    // ISA K-half selection per lane group
    auto compute = [&](int buf) {
        union FG { v16bf v; uint4 q[2]; };
        FG fa[2], fb[4];
#pragma unroll
        for (int mt = 0; mt < 2; ++mt) {
            const unsigned char* p = &shA[buf][(wm * 32 + mt * 16 + fr) * PITCH + fo];
            fa[mt].q[0] = *(const uint4*)p;
            fa[mt].q[1] = *(const uint4*)(p + 32);
        }
#pragma unroll
        for (int nt = 0; nt < 4; ++nt) {
            const unsigned char* p = &shB[buf][(wn * 64 + nt * 16 + fr) * PITCH + fo];
            fb[nt].q[0] = *(const uint4*)p;
            fb[nt].q[1] = *(const uint4*)(p + 32);
        }
#pragma unroll
        for (int mt = 0; mt < 2; ++mt)
#pragma unroll
            for (int nt = 0; nt < 4; ++nt)
                acc[mt][nt] = __builtin_amdgcn_wmma_f32_16x16x32_bf16(
                    false, fa[mt].v, false, fb[nt].v, (short)0, acc[mt][nt], false, false);
    };

    // prologue: tile 0
    if (issuer) tileB(0, 0);
    loadA(0);
    storeA(0);
    if (issuer) waitB();
    __syncthreads();

    const int NK = K_DIM / 32;   // 128 iterations, double-buffered
    for (int ki = 0; ki < NK; ++ki) {
        const int cb = ki & 1, nb = cb ^ 1;
        const int kk = (ki + 1) * 32;
        if (ki + 1 < NK) {
            if (issuer) tileB(kk, nb);    // B tile: TDM DMA global->LDS (bf16, L2-resident)
            loadA(kk);                    // A tile: fp32 -> regs
            __builtin_prefetch(agp + kk + 32, 0, 1);  // global_prefetch_b8 hint
        }
        compute(cb);                      // 12x ds_load_b128 + 8x v_wmma per wave
        if (ki + 1 < NK) {
            storeA(nb);                   // convert + ds_store (co-executes with WMMA)
            if (issuer) waitB();
        }
        __syncthreads();
    }

    // ---------------- epilogue: out = acc + bias + 2 * (h @ lora_b^T) ----------------
    const int g     = bm >> 5;            // 32 M-blocks per adapter group
    const int lmod  = lane & 15;
    const int lhalf = lane >> 4;
    const int mbase = m0 + wm * 32;
    const int nbase = n0 + wn * 64;
#pragma unroll
    for (int nt = 0; nt < 4; ++nt) {
        const int n = nbase + nt * 16 + lmod;
        const float bv = bias[n];
        const float4* bp = (const float4*)(lora_b + ((size_t)g * N_DIM + n) * 16);
        const float4 b0 = bp[0], b1 = bp[1], b2 = bp[2], b3 = bp[3];
#pragma unroll
        for (int mt = 0; mt < 2; ++mt) {
#pragma unroll
            for (int j = 0; j < 8; ++j) {
                const int m = mbase + mt * 16 + lhalf * 8 + j;
                const float4* hp = (const float4*)(hbuf + (size_t)m * 16);
                const float4 h0 = hp[0], h1 = hp[1], h2 = hp[2], h3 = hp[3];
                const float d =
                    h0.x * b0.x + h0.y * b0.y + h0.z * b0.z + h0.w * b0.w +
                    h1.x * b1.x + h1.y * b1.y + h1.z * b1.z + h1.w * b1.w +
                    h2.x * b2.x + h2.y * b2.y + h2.z * b2.z + h2.w * b2.w +
                    h3.x * b3.x + h3.y * b3.y + h3.z * b3.z + h3.w * b3.w;
                out[(size_t)m * N_DIM + n] = acc[mt][nt][j] + bv + 2.0f * d;
            }
        }
    }
}

// ---------------- launcher -----------------------------------------------------------------
extern "C" void kernel_launch(void* const* d_in, const int* in_sizes, int n_in,
                              void* d_out, int out_size, void* d_ws, size_t ws_size,
                              hipStream_t stream) {
    (void)in_sizes; (void)n_in; (void)out_size; (void)ws_size;
    const float* data = (const float*)d_in[0];   // [8,2048,4096]
    const float* W    = (const float*)d_in[1];   // [4096,4096]
    const float* bias = (const float*)d_in[2];   // [4096]
    const float* la   = (const float*)d_in[3];   // [4,16,4096]
    const float* lb   = (const float*)d_in[4];   // [4,4096,16]
    float* out = (float*)d_out;

    __bf16* wbf  = (__bf16*)d_ws;                                     // 33.5 MB
    float*  hbuf = (float*)((char*)d_ws + (size_t)N_DIM * K_DIM * 2); // 1 MB

    wconv_kernel <<<dim3((N_DIM * K_DIM) / (256 * 4)), 256, 0, stream>>>(W, wbf);
    lora_h_kernel<<<dim3((M_DIM * 16) / 256),          256, 0, stream>>>(data, la, hbuf);
    lora_gemm_kernel<<<dim3(N_DIM / 128, M_DIM / 128), 256, 0, stream>>>(
        data, bias, lb, wbf, hbuf, out);
}
